// NCNPredictor_39281770889759
// MI455X (gfx1250) — matrix operand order repaired and checked by reading, not compiled, not run on hardware
//
#include <hip/hip_runtime.h>
#include <hip/hip_bf16.h>

// NCN link predictor for MI455X (gfx1250).
// Dominant op: xcn = (ai (.) aj) @ x -> 34.4 GFLOP, 512MB minimal adjacency
// traffic (HBM floor ~22us @ 23.3 TB/s). ai*aj is exactly {0,1} -> lossless
// in f16 -> V_WMMA_F32_16X16X32_F16 with f32 accumulation everywhere.
//
// Stage-A decomposition: wave = (16-edge tile) x (K quarter of 2048); the 8
// waves of a workgroup share one K-quarter so they stream IDENTICAL B panels
// (16KB/K-step) -> WGP$ serves B, L2 only sees it once per block. Adjacency
// is read exactly once, non-temporally. Split-K partials combine via
// global_atomic_add_f32 (f32-exact reduction).

#define N_NODES 8192
#define E_EDGES 8192
#define IN_F    256
#define HID_F   256
#define KSPLIT  4
#define KCHUNK  (N_NODES / KSPLIT)   // 2048

typedef __attribute__((ext_vector_type(16))) _Float16 v16h;
typedef __attribute__((ext_vector_type(8)))  _Float16 v8h;
typedef __attribute__((ext_vector_type(8)))  float    v8f;
typedef __attribute__((ext_vector_type(4)))  float    v4f;

static __device__ __forceinline__ v4f ldg4_nt(const float* p) {
    return __builtin_nontemporal_load((const v4f*)p);
}

static __device__ __forceinline__ v8f wmma_f16(v16h a, v16h b, v8f c) {
    // D = A(16x32 f16) x B(32x16 f16) + C(16x16 f32)
    return __builtin_amdgcn_wmma_f32_16x16x32_f16(
        /*neg_a=*/false, a, /*neg_b=*/false, b,
        /*c_mod=*/(short)0, c, /*reuse_a=*/false, /*reuse_b=*/false);
}

// ---------------------------------------------------------------------------
// Prep kernels
// ---------------------------------------------------------------------------

// LDS-tiled transpose+downconvert: dst[c][r] = (f16) src[r][c].
// src is [R x C] f32 row-major, dst is [C x R] f16 row-major. 32x32 tiles,
// padded LDS row (33) to avoid bank conflicts; both sides fully coalesced.
__global__ void __launch_bounds__(256)
k_transpose_f16(const float* __restrict__ src, _Float16* __restrict__ dst,
                int R, int C) {
    __shared__ _Float16 tile[32][33];
    const int tb   = R >> 5;              // tiles along R
    const int bn   = blockIdx.x % tb;     // row-tile index
    const int bc   = blockIdx.x / tb;     // col-tile index
    const int r0   = bn * 32, c0 = bc * 32;
    const int lane = threadIdx.x & 31;
    const int ty   = threadIdx.x >> 5;    // 0..7
#pragma unroll
    for (int p = 0; p < 4; ++p) {
        int rr = p * 8 + ty;
        tile[rr][lane] = (_Float16)src[(size_t)(r0 + rr) * C + (c0 + lane)];
    }
    __syncthreads();
#pragma unroll
    for (int p = 0; p < 4; ++p) {
        int cc = p * 8 + ty;
        dst[(size_t)(c0 + cc) * R + (r0 + lane)] = tile[lane][cc];
    }
}

// xij[e][c] = (f16)( x[i_e][c] * x[j_e][c] )
__global__ void k_xij(const float* __restrict__ x, const int* __restrict__ tar,
                      _Float16* __restrict__ out) {
    int idx = blockIdx.x * blockDim.x + threadIdx.x;   // over E*IN
    int c = idx & (IN_F - 1);
    int e = idx >> 8;
    int i = tar[e];
    int j = tar[E_EDGES + e];
    out[idx] = (_Float16)(x[(size_t)i * IN_F + c] * x[(size_t)j * IN_F + c]);
}

__global__ void k_zero_f32(float* __restrict__ p) {
    int idx = blockIdx.x * blockDim.x + threadIdx.x;
    p[idx] = 0.f;
}

__global__ void k_f32_to_f16(const float* __restrict__ in,
                             _Float16* __restrict__ out) {
    int idx = blockIdx.x * blockDim.x + threadIdx.x;
    out[idx] = (_Float16)in[idx];
}

// ---------------------------------------------------------------------------
// Stage A: xcn += (ai (.) aj) @ x  over one K-quarter.
// One wave = one 16-edge tile x ALL 16 feature tiles; all waves of a block
// share the same K-quarter (kq = blockIdx&3) for WGP$-level B reuse.
// ---------------------------------------------------------------------------
__global__ void __launch_bounds__(256)
k_cn_gemm(const float* __restrict__ adj, const int* __restrict__ tar,
          const _Float16* __restrict__ xT, float* __restrict__ xcnF) {
    const int lane  = threadIdx.x & 31;
    const int wave  = threadIdx.x >> 5;          // 0..7
    const int kq    = blockIdx.x & 3;            // K quarter (shared in block)
    const int etile = (blockIdx.x >> 2) * 8 + wave;  // 0..511
    const int r     = lane & 15;         // A row / B col / C col within tile
    const int kb    = (lane >> 4) * 16;  // A-fragment K offset (contiguous k = 16h + s)
    const int kb2   = (lane >> 4) * 16;  // B-fragment K offset (contiguous k = 16h + s)

    const int  e  = etile * 16 + r;
    const long ri = tar[e];
    const long rj = tar[E_EDGES + e];
    const float* ai = adj + ri * (long)N_NODES;
    const float* aj = adj + rj * (long)N_NODES;

    // B base: row c = t*16 + (lane&15) of xT, K window at kb2
    const _Float16* bbase = xT + (size_t)r * N_NODES + kb2;

    v8f acc[16];
    const v8f vzero = {0.f, 0.f, 0.f, 0.f, 0.f, 0.f, 0.f, 0.f};
#pragma unroll
    for (int t = 0; t < 16; ++t) acc[t] = vzero;

    const int kbeg = kq * KCHUNK;
    const int kend = kbeg + KCHUNK;
    for (int k0 = kbeg; k0 < kend; k0 += 32) {
        __builtin_prefetch(ai + k0 + 256, 0, 1);
        __builtin_prefetch(aj + k0 + 256, 0, 1);
        // A fragment: rows of cn = ai*aj (exactly 0/1, f16-lossless)
        v4f i0 = ldg4_nt(ai + k0 + kb);       v4f j0 = ldg4_nt(aj + k0 + kb);
        v4f i1 = ldg4_nt(ai + k0 + kb + 4);   v4f j1 = ldg4_nt(aj + k0 + kb + 4);
        v4f i2 = ldg4_nt(ai + k0 + kb + 8);   v4f j2 = ldg4_nt(aj + k0 + kb + 8);
        v4f i3 = ldg4_nt(ai + k0 + kb + 12);  v4f j3 = ldg4_nt(aj + k0 + kb + 12);
        v16h a;
#pragma unroll
        for (int q = 0; q < 4; ++q) {
            a[q]      = (_Float16)(i0[q] * j0[q]);
            a[4 + q]  = (_Float16)(i1[q] * j1[q]);
            a[8 + q]  = (_Float16)(i2[q] * j2[q]);
            a[12 + q] = (_Float16)(i3[q] * j3[q]);
        }
#pragma unroll
        for (int t = 0; t < 16; ++t) {
            v16h b = *(const v16h*)(bbase + (size_t)t * (16 * N_NODES) + k0);
            acc[t] = wmma_f16(a, b, acc[t]);
        }
    }

    // C layout: col = lane&15, row = v + (lane>>4)*8. Split-K -> atomic add.
    const int mBase = (lane >> 4) * 8;
#pragma unroll
    for (int t = 0; t < 16; ++t) {
        int col = t * 16 + r;
#pragma unroll
        for (int v = 0; v < 8; ++v) {
            size_t o = (size_t)(etile * 16 + mBase + v) * HID_F + col;
            atomicAdd(&xcnF[o], acc[t][v]);
        }
    }
}

// ---------------------------------------------------------------------------
// Generic [E,256] @ [256,256] WMMA GEMM with fused bias / add / ReLU.
// A: f16 row-major. WT: f16 [out,in] (pre-transposed). One wave = 16 edges
// x 8 output tiles; K = 256 in 8 steps of 32.
// ---------------------------------------------------------------------------
__global__ void __launch_bounds__(256)
k_gemm256(const _Float16* __restrict__ A, const _Float16* __restrict__ WT,
          const float* __restrict__ bias, const float* __restrict__ addsrc,
          _Float16* __restrict__ outH, float* __restrict__ outF, int relu) {
    const int lane  = threadIdx.x & 31;
    const int wid   = blockIdx.x * (blockDim.x >> 5) + (threadIdx.x >> 5);
    const int etile = wid >> 1;
    const int nhalf = wid & 1;
    const int r     = lane & 15;
    const int kb    = (lane >> 4) * 16;
    const int kb2   = (lane >> 4) * 16;

    const _Float16* arow  = A + (size_t)(etile * 16 + r) * HID_F;
    const _Float16* bbase = WT + (size_t)(nhalf * 128 + r) * HID_F + kb2;

    v8f acc[8];
    const v8f vzero = {0.f, 0.f, 0.f, 0.f, 0.f, 0.f, 0.f, 0.f};
#pragma unroll
    for (int t = 0; t < 8; ++t) acc[t] = vzero;

#pragma unroll
    for (int k0 = 0; k0 < HID_F; k0 += 32) {
        v8h lo = *(const v8h*)(arow + k0 + kb);
        v8h hi = *(const v8h*)(arow + k0 + kb + 8);
        v16h a = __builtin_shufflevector(lo, hi, 0, 1, 2, 3, 4, 5, 6, 7,
                                         8, 9, 10, 11, 12, 13, 14, 15);
#pragma unroll
        for (int t = 0; t < 8; ++t) {
            v16h b = *(const v16h*)(bbase + (size_t)t * (16 * HID_F) + k0);
            acc[t] = wmma_f16(a, b, acc[t]);
        }
    }

    const int mBase = (lane >> 4) * 8;
#pragma unroll
    for (int t = 0; t < 8; ++t) {
        int col = nhalf * 128 + t * 16 + r;
        float bc = bias ? bias[col] : 0.f;
#pragma unroll
        for (int v = 0; v < 8; ++v) {
            size_t o = (size_t)(etile * 16 + mBase + v) * HID_F + col;
            float val = acc[t][v] + bc;
            if (addsrc) val += addsrc[o];
            if (relu) val = fmaxf(val, 0.f);
            if (outH) outH[o] = (_Float16)val;
            else      outF[o] = val;
        }
    }
}

// out[e] = dot(g2[e], Wout) + bout
__global__ void k_head(const _Float16* __restrict__ g2,
                       const float* __restrict__ Wout,
                       const float* __restrict__ bout,
                       float* __restrict__ out) {
    int e = blockIdx.x * blockDim.x + threadIdx.x;
    if (e >= E_EDGES) return;
    const _Float16* row = g2 + (size_t)e * HID_F;
    float s = 0.f;
#pragma unroll 8
    for (int c = 0; c < HID_F; ++c) s += (float)row[c] * Wout[c];
    out[e] = s + bout[0];
}

// ---------------------------------------------------------------------------
extern "C" void kernel_launch(void* const* d_in, const int* in_sizes, int n_in,
                              void* d_out, int out_size, void* d_ws,
                              size_t ws_size, hipStream_t stream) {
    const float* x    = (const float*)d_in[0];
    const float* adj  = (const float*)d_in[1];
    const int*   tar  = (const int*)d_in[2];
    const float* W1c  = (const float*)d_in[3];  const float* b1c = (const float*)d_in[4];
    const float* W2c  = (const float*)d_in[5];  const float* b2c = (const float*)d_in[6];
    const float* W1j  = (const float*)d_in[7];  const float* b1j = (const float*)d_in[8];
    const float* W2j  = (const float*)d_in[9];  const float* b2j = (const float*)d_in[10];
    const float* Wl1  = (const float*)d_in[11]; const float* bl1 = (const float*)d_in[12];
    const float* Wl2  = (const float*)d_in[13]; const float* bl2 = (const float*)d_in[14];
    const float* Wout = (const float*)d_in[15]; const float* bout = (const float*)d_in[16];
    float* out = (float*)d_out;

    // Workspace layout (bytes). SZ = 4MB per f16 [E,256] buffer.
    char* ws = (char*)d_ws;
    const size_t SZ = (size_t)E_EDGES * HID_F * sizeof(_Float16);  // 4 MiB
    _Float16* xT   = (_Float16*)(ws + 0 * SZ);  // N*IN f16 = 4 MiB
    _Float16* xij  = (_Float16*)(ws + 1 * SZ);
    _Float16* xcn  = (_Float16*)(ws + 2 * SZ);
    _Float16* h1c  = (_Float16*)(ws + 3 * SZ);
    _Float16* h1j  = (_Float16*)(ws + 4 * SZ);
    float*    sF   = (float*)   (ws + 5 * SZ);  // 8 MiB f32
    // aliases of dead buffers (sequential kernels on one stream):
    _Float16* hb  = h1c;   // h   overwrites h1c (dead after sF)
    _Float16* g1  = h1j;   // g1  overwrites h1j (dead after h)
    _Float16* g2  = xij;   // g2  overwrites xij (dead after h1j)
    char* wp = ws + 5 * SZ + (size_t)E_EDGES * HID_F * sizeof(float);
    const size_t WSZ = (size_t)HID_F * HID_F * sizeof(_Float16);   // 128 KiB
    _Float16* W1cT = (_Float16*)(wp + 0 * WSZ);
    _Float16* W2cT = (_Float16*)(wp + 1 * WSZ);
    _Float16* W1jT = (_Float16*)(wp + 2 * WSZ);
    _Float16* W2jT = (_Float16*)(wp + 3 * WSZ);
    _Float16* Wl1T = (_Float16*)(wp + 4 * WSZ);
    _Float16* Wl2T = (_Float16*)(wp + 5 * WSZ);
    float*    xcnF = (float*)   (wp + 6 * WSZ); // 8 MiB split-K accumulator

    const int TB = 256;
    const int TR_X = (N_NODES / 32) * (IN_F / 32);   // 2048 tiles
    const int TR_W = (HID_F / 32) * (HID_F / 32);    // 64 tiles
    // prep (LDS-tiled transposes: coalesced + ds traffic)
    k_transpose_f16<<<TR_X, TB, 0, stream>>>(x, xT, N_NODES, IN_F);
    k_xij<<<(E_EDGES * IN_F) / TB, TB, 0, stream>>>(x, tar, xij);
    k_zero_f32<<<(E_EDGES * HID_F) / TB, TB, 0, stream>>>(xcnF);
    k_transpose_f16<<<TR_W, TB, 0, stream>>>(W1c, W1cT, HID_F, HID_F);
    k_transpose_f16<<<TR_W, TB, 0, stream>>>(W2c, W2cT, HID_F, HID_F);
    k_transpose_f16<<<TR_W, TB, 0, stream>>>(W1j, W1jT, HID_F, HID_F);
    k_transpose_f16<<<TR_W, TB, 0, stream>>>(W2j, W2jT, HID_F, HID_F);
    k_transpose_f16<<<TR_W, TB, 0, stream>>>(Wl1, Wl1T, HID_F, HID_F);
    k_transpose_f16<<<TR_W, TB, 0, stream>>>(Wl2, Wl2T, HID_F, HID_F);

    // Stage A: 2048 waves = 256 blocks x 8 waves, split-K atomics
    k_cn_gemm<<<256, TB, 0, stream>>>(adj, tar, xT, xcnF);
    k_f32_to_f16<<<(E_EDGES * HID_F) / TB, TB, 0, stream>>>(xcnF, xcn);

    // MLP chain (all WMMA):
    // h1c = relu(xcn @ W1c + b1c)
    k_gemm256<<<128, TB, 0, stream>>>(xcn, W1cT, b1c, nullptr, h1c, nullptr, 1);
    // h1j = relu(xij @ W1j + b1j)
    k_gemm256<<<128, TB, 0, stream>>>(xij, W1jT, b1j, nullptr, h1j, nullptr, 1);
    // sF = h1c @ W2c + b2c            (f32, no relu)
    k_gemm256<<<128, TB, 0, stream>>>(h1c, W2cT, b2c, nullptr, nullptr, sF, 0);
    // h = (h1j @ W2j + b2j) + sF      (beta = 1)
    k_gemm256<<<128, TB, 0, stream>>>(h1j, W2jT, b2j, sF, hb, nullptr, 0);
    // g1 = relu(h @ Wl1 + bl1)
    k_gemm256<<<128, TB, 0, stream>>>(hb, Wl1T, bl1, nullptr, g1, nullptr, 1);
    // g2 = relu(g1 @ Wl2 + bl2)
    k_gemm256<<<128, TB, 0, stream>>>(g1, Wl2T, bl2, nullptr, g2, nullptr, 1);
    // out = g2 @ Wout + bout
    k_head<<<E_EDGES / TB, TB, 0, stream>>>(g2, Wout, bout, out);
}
